// HierarchicalAttentionNetwork_4105988735594
// MI455X (gfx1250) — matrix operand order repaired and checked by reading, not compiled, not run on hardware
//
#include <hip/hip_runtime.h>
#include <hip/hip_bf16.h>
#include <math.h>

typedef _Float16 f16;
typedef __attribute__((ext_vector_type(16))) _Float16 v16h;
typedef __attribute__((ext_vector_type(8)))  float    v8f;

#define IN_DIM 256
#define HID    128
#define H2     64
#define HEADS  4
#define DH     32
#define BB     8
#define SS     2048
#define NMEAS  16

// ---------------------------------------------------------------------------
// f32 -> f16 conversion
// ---------------------------------------------------------------------------
__global__ void cvt_f16_kernel(const float* __restrict__ in, f16* __restrict__ out, int n) {
  int i = blockIdx.x * blockDim.x + threadIdx.x;
  if (i < n) out[i] = (f16)in[i];
}

// ---------------------------------------------------------------------------
// Extract V from packed QKV (B,S,384) into transposed (B,HEADS,DH,S) layout so
// the flash-attention P@V B-matrix loads become contiguous 128-bit loads.
// Output-indexed for coalesced stores.
// ---------------------------------------------------------------------------
__global__ void v_transpose_kernel(const f16* __restrict__ qkv, f16* __restrict__ vt, int S) {
  int idx = blockIdx.x * blockDim.x + threadIdx.x;     // over B*HEADS*DH*S
  if (idx >= BB * HEADS * DH * S) return;
  int s = idx % S;
  int t = idx / S;
  int d = t % DH;  t /= DH;
  int h = t % HEADS;
  int b = t / HEADS;
  vt[idx] = qkv[((size_t)b * S + s) * (3 * HID) + 2 * HID + h * DH + d];
}

// ---------------------------------------------------------------------------
// WMMA GEMM: C(MxN) = A(MxK) @ W(NxK)^T + b1 + b2   (f16 in, f32 acc)
// One wave per 16x16 C tile. K multiple of 32, M,N multiples of 16.
// A layout per CDNA5 16-bit A (16x32): lanes 0-15 rows, K halves split by group.
// B layout: lane = N column, K striped (contiguous along W rows => pair loads,
// which the compiler merges into global_load_b128).
// ---------------------------------------------------------------------------
__global__ __launch_bounds__(32) void wmma_gemm_bt(
    const f16* __restrict__ A, const f16* __restrict__ W,
    const float* __restrict__ b1, const float* __restrict__ b2,
    float* __restrict__ Cf, f16* __restrict__ Ch,
    int M, int N, int K)
{
  const int mt   = blockIdx.x;
  const int nt   = blockIdx.y;
  const int lane = threadIdx.x;
  const int g    = lane >> 4;      // lane group (K-half select)
  const int l16  = lane & 15;
  const int mrow = mt * 16 + l16;  // A row this lane loads
  const int ncol = nt * 16 + l16;  // W row (output column) this lane loads

  v8f acc = {};
  for (int kb = 0; kb < K; kb += 32) {
    v16h a, b;
    unsigned int* aw = reinterpret_cast<unsigned int*>(&a);
    unsigned int* bw = reinterpret_cast<unsigned int*>(&b);
    const f16* ap = A + (size_t)mrow * K + kb + g * 8;
#pragma unroll
    for (int j = 0; j < 4; ++j) aw[j]     = *reinterpret_cast<const unsigned int*>(ap + 2 * j);
#pragma unroll
    for (int j = 0; j < 4; ++j) aw[4 + j] = *reinterpret_cast<const unsigned int*>(ap + 16 + 2 * j);
    const f16* bp = W + (size_t)ncol * K + kb + g * 8;
#pragma unroll
    for (int j = 0; j < 8; ++j) bw[j]     = *reinterpret_cast<const unsigned int*>(bp + (j < 4 ? 2 * j : 16 + 2 * (j - 4)));
    acc = __builtin_amdgcn_wmma_f32_16x16x32_f16(false, a, false, b, (short)0, acc, false, false);
  }

  float bias = 0.0f;
  if (b1) bias += b1[ncol];
  if (b2) bias += b2[ncol];

#pragma unroll
  for (int v = 0; v < 8; ++v) {
    int row = mt * 16 + v + 8 * g;   // C layout: lane group selects M half
    float val = acc[v] + bias;
    if (Cf) Cf[(size_t)row * N + ncol] = val;
    if (Ch) Ch[(size_t)row * N + ncol] = (f16)val;
  }
}

// ---------------------------------------------------------------------------
// Sequential LSTM recurrence (one direction). Input projection G already
// contains x@Wih^T + bih + bhh for every timestep. One block per batch.
// 256 threads: each computes one gate pre-activation (dot with h, 64 MACs),
// threads 0..63 then update c/h and write the hidden state.
// ---------------------------------------------------------------------------
__global__ __launch_bounds__(256) void lstm_rec(
    const float* __restrict__ G,    // (B,S,4*H2) with bias folded in
    const float* __restrict__ Whh,  // (4*H2, H2) row-major f32
    float* __restrict__ Hout,       // (B,S,HID), this direction writes [outOff, outOff+H2)
    int S, int reverse, int outOff)
{
  __shared__ float h[H2];
  __shared__ float c[H2];
  __shared__ float gates[4 * H2];

  const int b = blockIdx.x;
  const int n = threadIdx.x;
  if (n < H2) { h[n] = 0.0f; c[n] = 0.0f; }
  __syncthreads();

  float w[H2];
#pragma unroll
  for (int k = 0; k < H2; ++k) w[k] = Whh[n * H2 + k];

  for (int step = 0; step < S; ++step) {
    const int t = reverse ? (S - 1 - step) : step;
    const float* gp = G + ((size_t)b * S + t) * (4 * H2);
    float acc = gp[n];
#pragma unroll 8
    for (int k = 0; k < H2; ++k) acc += w[k] * h[k];
    gates[n] = acc;
    __syncthreads();
    if (n < H2) {
      float ig = 1.0f / (1.0f + __expf(-gates[n]));
      float fg = 1.0f / (1.0f + __expf(-gates[H2 + n]));
      float gg = tanhf(gates[2 * H2 + n]);
      float og = 1.0f / (1.0f + __expf(-gates[3 * H2 + n]));
      float cn = fg * c[n] + ig * gg;
      c[n] = cn;
      float hn = og * tanhf(cn);
      h[n] = hn;
      Hout[((size_t)b * S + t) * HID + outOff + n] = hn;
    }
    __syncthreads();
  }
}

// ---------------------------------------------------------------------------
// Flash attention, one wave per (batch, head, 16-query tile).
// qkv: (B,S,384) f16 holds Q (col 0) and K (col 128); V comes pre-transposed
// as Vt (B,HEADS,DH,S) so its B-matrix loads are contiguous. dh=32 == one
// WMMA K-step. Online softmax; P is transposed through a 16x32 LDS tile to
// re-enter the A-matrix layout.
// ---------------------------------------------------------------------------
__global__ __launch_bounds__(32) void flash_attn(
    const f16* __restrict__ qkv, const f16* __restrict__ Vt,
    f16* __restrict__ out, int S, float scale)
{
  __shared__ f16 P[16 * 32];

  const int bh   = blockIdx.x;
  const int b    = bh / HEADS;
  const int h    = bh % HEADS;
  const int qt   = blockIdx.y;
  const int lane = threadIdx.x;
  const int g    = lane >> 4;
  const int l16  = lane & 15;
  const int ld   = 3 * HID;     // 384

  // Q tile in A layout
  v16h qa;
  {
    unsigned int* aw = reinterpret_cast<unsigned int*>(&qa);
    const f16* qp = qkv + (size_t)(b * S + qt * 16 + l16) * ld + h * DH + g * 8;
#pragma unroll
    for (int j = 0; j < 4; ++j) aw[j]     = *reinterpret_cast<const unsigned int*>(qp + 2 * j);
#pragma unroll
    for (int j = 0; j < 4; ++j) aw[4 + j] = *reinterpret_cast<const unsigned int*>(qp + 16 + 2 * j);
  }

  v8f o0 = {}, o1 = {};
  float m[8], l[8];
#pragma unroll
  for (int v = 0; v < 8; ++v) { m[v] = -1.0e30f; l[v] = 0.0f; }

  for (int kc = 0; kc < S; kc += 32) {
    v8f s0, s1;
    const v8f zacc = {};
#pragma unroll
    for (int jn = 0; jn < 2; ++jn) {
      const int key = kc + jn * 16 + l16;
      const bool ok = key < S;
      v16h kb;
      unsigned int* bw = reinterpret_cast<unsigned int*>(&kb);
      const f16* kp = qkv + (size_t)(b * S + key) * ld + HID + h * DH + g * 8;
#pragma unroll
      for (int j = 0; j < 8; ++j)
        bw[j] = ok ? *reinterpret_cast<const unsigned int*>(kp + (j < 4 ? 2 * j : 16 + 2 * (j - 4))) : 0u;
      v8f s = __builtin_amdgcn_wmma_f32_16x16x32_f16(false, qa, false, kb, (short)0, zacc, false, false);
#pragma unroll
      for (int v = 0; v < 8; ++v) s[v] = ok ? s[v] * scale : -1.0e30f;
      if (jn == 0) s0 = s; else s1 = s;
    }

    // online softmax update per row (rows live on 16-lane groups)
#pragma unroll
    for (int v = 0; v < 8; ++v) {
      float cm = fmaxf(s0[v], s1[v]);
      cm = fmaxf(cm, __shfl_xor(cm, 1, 32));
      cm = fmaxf(cm, __shfl_xor(cm, 2, 32));
      cm = fmaxf(cm, __shfl_xor(cm, 4, 32));
      cm = fmaxf(cm, __shfl_xor(cm, 8, 32));
      float mn    = fmaxf(m[v], cm);
      float alpha = __expf(m[v] - mn);
      float p0    = __expf(s0[v] - mn);
      float p1    = __expf(s1[v] - mn);
      float rs = p0 + p1;
      rs += __shfl_xor(rs, 1, 32);
      rs += __shfl_xor(rs, 2, 32);
      rs += __shfl_xor(rs, 4, 32);
      rs += __shfl_xor(rs, 8, 32);
      l[v] = l[v] * alpha + rs;
      m[v] = mn;
      o0[v] *= alpha;
      o1[v] *= alpha;
      const int row = v + 8 * g;
      P[row * 32 + l16]      = (f16)p0;
      P[row * 32 + 16 + l16] = (f16)p1;
    }
    __syncthreads();

    // reload P in A layout
    v16h pa;
    {
      unsigned int* aw = reinterpret_cast<unsigned int*>(&pa);
      const f16* pp = P + l16 * 32 + g * 16;
#pragma unroll
      for (int j = 0; j < 4; ++j) aw[j]     = *reinterpret_cast<const unsigned int*>(pp + 2 * j);
#pragma unroll
      for (int j = 0; j < 4; ++j) aw[4 + j] = *reinterpret_cast<const unsigned int*>(pp + 8 + 2 * j);
    }

    // V B-tiles from transposed layout: row = dh column, contiguous keys.
    // Key run [kc+16g, kc+16g+15] has uniform validity since S % 16 == 0.
    const bool okv = (kc + 16 * g) < S;
#pragma unroll
    for (int jv = 0; jv < 2; ++jv) {
      v16h vb;
      unsigned int* bw = reinterpret_cast<unsigned int*>(&vb);
      const f16* vp = Vt + (((size_t)(b * HEADS + h)) * DH + jv * 16 + l16) * S + kc + 16 * g;
#pragma unroll
      for (int j = 0; j < 8; ++j)
        bw[j] = okv ? *reinterpret_cast<const unsigned int*>(vp + 2 * j) : 0u;
      if (jv == 0)
        o0 = __builtin_amdgcn_wmma_f32_16x16x32_f16(false, pa, false, vb, (short)0, o0, false, false);
      else
        o1 = __builtin_amdgcn_wmma_f32_16x16x32_f16(false, pa, false, vb, (short)0, o1, false, false);
    }
    __syncthreads();
  }

#pragma unroll
  for (int v = 0; v < 8; ++v) {
    const float inv = 1.0f / l[v];
    const int row = qt * 16 + v + 8 * g;
    out[(size_t)(b * S + row) * HID + h * DH + l16]      = (f16)(o0[v] * inv);
    out[(size_t)(b * S + row) * HID + h * DH + 16 + l16] = (f16)(o1[v] * inv);
  }
}

// ---------------------------------------------------------------------------
// Segment mean over the sequence dim. One block per (batch, segment),
// 128 threads = feature dim. count clamped to >=1 like the reference.
// ---------------------------------------------------------------------------
__global__ __launch_bounds__(128) void seg_mean_kernel(
    const float* __restrict__ feat, const int* __restrict__ ids,
    f16* __restrict__ out, int S, int nseg)
{
  const int b   = blockIdx.x / nseg;
  const int seg = blockIdx.x % nseg;
  const int f   = threadIdx.x;
  float sum = 0.0f;
  int cnt = 0;
  for (int s = 0; s < S; ++s) {
    if (ids[b * S + s] == seg) {
      sum += feat[((size_t)b * S + s) * HID + f];
      cnt++;
    }
  }
  const float mean = sum / (float)(cnt > 0 ? cnt : 1);
  out[(size_t)(b * nseg + seg) * HID + f] = (f16)mean;
}

// ---------------------------------------------------------------------------
// Final: out[b] = measure[b, last, :] . fcW + fcb
// ---------------------------------------------------------------------------
__global__ __launch_bounds__(32) void final_fc(
    const float* __restrict__ mout, const float* __restrict__ fcW,
    const float* __restrict__ fcb, float* __restrict__ out)
{
  const int b = threadIdx.x;
  if (b < BB) {
    const float* r = mout + ((size_t)b * NMEAS + (NMEAS - 1)) * HID;
    float s = 0.0f;
    for (int f = 0; f < HID; ++f) s += r[f] * fcW[f];
    out[b] = s + fcb[0];
  }
}

// ---------------------------------------------------------------------------
// Host orchestration
// ---------------------------------------------------------------------------
extern "C" void kernel_launch(void* const* d_in, const int* in_sizes, int n_in,
                              void* d_out, int out_size, void* d_ws, size_t ws_size,
                              hipStream_t stream) {
  (void)in_sizes; (void)n_in; (void)out_size; (void)ws_size;

  const float* x           = (const float*)d_in[0];
  const int*   measure_pos = (const int*)d_in[4];
  // note BiLSTM params
  const float* nWih_f = (const float*)d_in[5];
  const float* nWhh_f = (const float*)d_in[6];
  const float* nbih_f = (const float*)d_in[7];
  const float* nbhh_f = (const float*)d_in[8];
  const float* nWih_r = (const float*)d_in[9];
  const float* nWhh_r = (const float*)d_in[10];
  const float* nbih_r = (const float*)d_in[11];
  const float* nbhh_r = (const float*)d_in[12];
  // note attention params
  const float* nWqkv = (const float*)d_in[13];
  const float* nbqkv = (const float*)d_in[14];
  const float* nWo   = (const float*)d_in[15];
  const float* nbo   = (const float*)d_in[16];
  // (voice: 17-28, beat: 29-40 — dead code in the reference, skipped)
  // measure BiLSTM params
  const float* mWih_f = (const float*)d_in[41];
  const float* mWhh_f = (const float*)d_in[42];
  const float* mbih_f = (const float*)d_in[43];
  const float* mbhh_f = (const float*)d_in[44];
  const float* mWih_r = (const float*)d_in[45];
  const float* mWhh_r = (const float*)d_in[46];
  const float* mbih_r = (const float*)d_in[47];
  const float* mbhh_r = (const float*)d_in[48];
  // measure attention params
  const float* mWqkv = (const float*)d_in[49];
  const float* mbqkv = (const float*)d_in[50];
  const float* mWo   = (const float*)d_in[51];
  const float* mbo   = (const float*)d_in[52];
  const float* fcW   = (const float*)d_in[53];
  const float* fcb   = (const float*)d_in[54];

  char* base = (char*)d_ws;
  size_t off = 0;
  auto alloc = [&](size_t bytes) -> void* {
    off = (off + 255) & ~(size_t)255;
    void* p = base + off;
    off += bytes;
    return p;
  };

  const int MN = BB * SS;        // 16384 note rows
  const int MM = BB * NMEAS;     // 128 measure rows

  f16* x16       = (f16*)alloc((size_t)MN * IN_DIM * sizeof(f16));
  f16* wih_f16   = (f16*)alloc((size_t)4 * H2 * IN_DIM * sizeof(f16));
  f16* wih_r16   = (f16*)alloc((size_t)4 * H2 * IN_DIM * sizeof(f16));
  f16* wqkv16    = (f16*)alloc((size_t)3 * HID * HID * sizeof(f16));
  f16* wo16      = (f16*)alloc((size_t)HID * HID * sizeof(f16));
  f16* mwih_f16  = (f16*)alloc((size_t)4 * H2 * HID * sizeof(f16));
  f16* mwih_r16  = (f16*)alloc((size_t)4 * H2 * HID * sizeof(f16));
  f16* mwqkv16   = (f16*)alloc((size_t)3 * HID * HID * sizeof(f16));
  f16* mwo16     = (f16*)alloc((size_t)HID * HID * sizeof(f16));
  float* Gf      = (float*)alloc((size_t)MN * 4 * H2 * sizeof(float));
  float* Gr      = (float*)alloc((size_t)MN * 4 * H2 * sizeof(float));
  float* note_h  = (float*)alloc((size_t)MN * HID * sizeof(float));
  f16* note_h16  = (f16*)alloc((size_t)MN * HID * sizeof(f16));
  f16* qkv16     = (f16*)alloc((size_t)MN * 3 * HID * sizeof(f16));
  f16* vt16      = (f16*)alloc((size_t)MN * HID * sizeof(f16));
  f16* attout16  = (f16*)alloc((size_t)MN * HID * sizeof(f16));
  float* note_out= (float*)alloc((size_t)MN * HID * sizeof(float));
  f16* mseg16    = (f16*)alloc((size_t)MM * HID * sizeof(f16));
  float* mGf     = (float*)alloc((size_t)MM * 4 * H2 * sizeof(float));
  float* mGr     = (float*)alloc((size_t)MM * 4 * H2 * sizeof(float));
  float* mh      = (float*)alloc((size_t)MM * HID * sizeof(float));
  f16* mh16      = (f16*)alloc((size_t)MM * HID * sizeof(f16));
  f16* mqkv16    = (f16*)alloc((size_t)MM * 3 * HID * sizeof(f16));
  f16* mvt16     = (f16*)alloc((size_t)MM * HID * sizeof(f16));
  f16* mattout16 = (f16*)alloc((size_t)MM * HID * sizeof(f16));
  float* mout    = (float*)alloc((size_t)MM * HID * sizeof(float));

  auto cvt = [&](const float* src, f16* dst, int n) {
    cvt_f16_kernel<<<(n + 255) / 256, 256, 0, stream>>>(src, dst, n);
  };

  // --- precision staging -------------------------------------------------
  cvt(x,      x16,      MN * IN_DIM);
  cvt(nWih_f, wih_f16,  4 * H2 * IN_DIM);
  cvt(nWih_r, wih_r16,  4 * H2 * IN_DIM);
  cvt(nWqkv,  wqkv16,   3 * HID * HID);
  cvt(nWo,    wo16,     HID * HID);
  cvt(mWih_f, mwih_f16, 4 * H2 * HID);
  cvt(mWih_r, mwih_r16, 4 * H2 * HID);
  cvt(mWqkv,  mwqkv16,  3 * HID * HID);
  cvt(mWo,    mwo16,    HID * HID);

  // --- note BiLSTM -------------------------------------------------------
  // input projections (all timesteps at once, bias folded)
  wmma_gemm_bt<<<dim3(MN / 16, (4 * H2) / 16), 32, 0, stream>>>(
      x16, wih_f16, nbih_f, nbhh_f, Gf, nullptr, MN, 4 * H2, IN_DIM);
  wmma_gemm_bt<<<dim3(MN / 16, (4 * H2) / 16), 32, 0, stream>>>(
      x16, wih_r16, nbih_r, nbhh_r, Gr, nullptr, MN, 4 * H2, IN_DIM);
  // sequential recurrences (latency-bound part)
  lstm_rec<<<BB, 256, 0, stream>>>(Gf, nWhh_f, note_h, SS, 0, 0);
  lstm_rec<<<BB, 256, 0, stream>>>(Gr, nWhh_r, note_h, SS, 1, H2);
  cvt(note_h, note_h16, MN * HID);

  // --- note MHA ----------------------------------------------------------
  wmma_gemm_bt<<<dim3(MN / 16, (3 * HID) / 16), 32, 0, stream>>>(
      note_h16, wqkv16, nbqkv, nullptr, nullptr, qkv16, MN, 3 * HID, HID);
  v_transpose_kernel<<<(MN * HID + 255) / 256, 256, 0, stream>>>(qkv16, vt16, SS);
  flash_attn<<<dim3(BB * HEADS, SS / 16), 32, 0, stream>>>(
      qkv16, vt16, attout16, SS, 0.17677669529663687f /* 1/sqrt(32) */);
  wmma_gemm_bt<<<dim3(MN / 16, HID / 16), 32, 0, stream>>>(
      attout16, wo16, nbo, nullptr, note_out, nullptr, MN, HID, HID);

  // --- measure path (voice/beat are dead in the reference) ---------------
  seg_mean_kernel<<<BB * NMEAS, HID, 0, stream>>>(
      note_out, measure_pos, mseg16, SS, NMEAS);

  wmma_gemm_bt<<<dim3(MM / 16, (4 * H2) / 16), 32, 0, stream>>>(
      mseg16, mwih_f16, mbih_f, mbhh_f, mGf, nullptr, MM, 4 * H2, HID);
  wmma_gemm_bt<<<dim3(MM / 16, (4 * H2) / 16), 32, 0, stream>>>(
      mseg16, mwih_r16, mbih_r, mbhh_r, mGr, nullptr, MM, 4 * H2, HID);
  lstm_rec<<<BB, 256, 0, stream>>>(mGf, mWhh_f, mh, NMEAS, 0, 0);
  lstm_rec<<<BB, 256, 0, stream>>>(mGr, mWhh_r, mh, NMEAS, 1, H2);
  cvt(mh, mh16, MM * HID);

  wmma_gemm_bt<<<dim3(MM / 16, (3 * HID) / 16), 32, 0, stream>>>(
      mh16, mwqkv16, mbqkv, nullptr, nullptr, mqkv16, MM, 3 * HID, HID);
  v_transpose_kernel<<<(MM * HID + 255) / 256, 256, 0, stream>>>(mqkv16, mvt16, NMEAS);
  flash_attn<<<dim3(BB * HEADS, NMEAS / 16), 32, 0, stream>>>(
      mqkv16, mvt16, mattout16, NMEAS, 0.17677669529663687f);
  wmma_gemm_bt<<<dim3(MM / 16, HID / 16), 32, 0, stream>>>(
      mattout16, mwo16, mbo, nullptr, mout, nullptr, MM, HID, HID);

  // --- head --------------------------------------------------------------
  final_fc<<<1, 32, 0, stream>>>(mout, fcW, fcb, (float*)d_out);
}
